// EmbodiedMAEPointCloudEmbeddings_55241869361621
// MI455X (gfx1250) — matrix operand order repaired and checked by reading, not compiled, not run on hardware
//
#include <hip/hip_runtime.h>
#include <hip/hip_bf16.h>
#include <math.h>

// ---------------- problem constants ----------------
static constexpr int BB   = 32;    // batch
static constexpr int NN   = 8192;  // points
static constexpr int CC   = 512;   // centers
static constexpr int KK   = 32;    // knn
static constexpr int HH   = 768;   // hidden
static constexpr int BC   = BB * CC;      // 16384 rows
static constexpr float LN_EPS = 1e-5f;

// ---------------- vector types for WMMA ----------------
typedef __attribute__((ext_vector_type(16))) __bf16 v16bf;
typedef __attribute__((ext_vector_type(8)))  __bf16 v8bf;
typedef __attribute__((ext_vector_type(8)))  float  v8f;

__device__ __forceinline__ v16bf make16(v8bf a, v8bf b) {
  v16bf r;
#pragma unroll
  for (int i = 0; i < 8; ++i) { r[i] = a[i]; r[i + 8] = b[i]; }
  return r;
}

// A-fragment (16x32 bf16, row-major source with row stride `ld`):
// lane16 = M, half selects k-group; elements 0..7 -> K = half*8 + e,
// elements 8..15 -> K = 16 + half*8 + (e-8). Two contiguous 8-elem runs.
__device__ __forceinline__ v16bf load_a_frag(const __bf16* h, int ld, int mbase,
                                             int kbase, int lane) {
  int lane16 = lane & 15, half = lane >> 4;
  const __bf16* p = h + (size_t)(mbase + lane16) * ld + kbase + half * 8;
  v8bf lo = *(const v8bf*)p;
  v8bf hi = *(const v8bf*)(p + 16);
  return make16(lo, hi);
}

// B-fragment (32x16 bf16) from N-major (transposed) storage BT[N][Kd]:
// lane16 = N, half selects k-group; elements 0..7 -> K = half*8 + e, 8..15 -> K = 16 + half*8 + (e-8).
__device__ __forceinline__ v16bf load_b_frag(const __bf16* wt, int ldk, int nbase,
                                             int kbase, int lane) {
  int lane16 = lane & 15, half = lane >> 4;
  const __bf16* p = wt + (size_t)(nbase + lane16) * ldk + kbase + half * 8;
  v8bf lo = *(const v8bf*)p;
  v8bf hi = *(const v8bf*)(p + 16);
  return make16(lo, hi);
}

__device__ __forceinline__ float gelu_tanh(float x) {
  float x3 = x * x * x;
  return 0.5f * x * (1.0f + tanhf(0.7978845608028654f * (x + 0.044715f * x3)));
}

// ---------------- CDNA5 async global->LDS copy (ASYNCcnt) ----------------
// lds32: wave-relative LDS byte address (low 32 bits of a generic pointer to
// a __shared__ object; flat-LDS addresses keep the offset in bits [31:0]).
__device__ __forceinline__ void async_b128(unsigned lds32, const void* gaddr) {
  asm volatile("global_load_async_to_lds_b128 %0, %1, off"
               :: "v"(lds32), "v"(gaddr) : "memory");
}
__device__ __forceinline__ void wait_async0() {
  asm volatile("s_wait_asynccnt 0x0" ::: "memory");
}
__device__ __forceinline__ unsigned lds_off32(const void* p) {
  return (unsigned)(unsigned long long)p;
}

// ============================================================
// Kernel A: farthest point sampling. 1 block / batch, 256 thr.
// ============================================================
__global__ void fps_kernel(const float* __restrict__ pc, float* __restrict__ centers) {
  const int b = blockIdx.x;
  const int t = threadIdx.x;
  const float* x = pc + (size_t)b * NN * 3;

  float px[32], py[32], pz[32], dd[32];
#pragma unroll
  for (int j = 0; j < 32; ++j) {
    int p = j * 256 + t;
    px[j] = x[p * 3 + 0]; py[j] = x[p * 3 + 1]; pz[j] = x[p * 3 + 2];
    dd[j] = 1e10f;
  }

  __shared__ float rv[256];
  __shared__ int   ri[256];
  __shared__ float cpt[3];
  __shared__ int   s_sel;
  if (t == 0) s_sel = 0;
  __syncthreads();

  for (int k = 0; k < CC; ++k) {
    int sel = s_sel;
    if (t == (sel & 255)) {
      int j = sel >> 8;
      cpt[0] = px[j]; cpt[1] = py[j]; cpt[2] = pz[j];
    }
    __syncthreads();
    float cx = cpt[0], cy = cpt[1], cz = cpt[2];
    if (t == 0) {
      float* co = centers + ((size_t)b * CC + k) * 3;
      co[0] = cx; co[1] = cy; co[2] = cz;
    }
    float bestv = -1.0f; int besti = 0;
#pragma unroll
    for (int j = 0; j < 32; ++j) {
      float dx = px[j] - cx, dy = py[j] - cy, dz = pz[j] - cz;
      float d2 = dx * dx + dy * dy + dz * dz;
      dd[j] = fminf(dd[j], d2);
      if (dd[j] > bestv) { bestv = dd[j]; besti = j * 256 + t; }
    }
    rv[t] = bestv; ri[t] = besti;
    __syncthreads();
    for (int s = 128; s > 0; s >>= 1) {
      if (t < s) {
        float v2 = rv[t + s]; int i2 = ri[t + s];
        if (v2 > rv[t] || (v2 == rv[t] && i2 < ri[t])) { rv[t] = v2; ri[t] = i2; }
      }
      __syncthreads();
    }
    if (t == 0) s_sel = ri[0];
    __syncthreads();
  }
}

// ============================================================
// Kernel B: KNN (top-32 nearest, ascending order) + normed out.
// 1 block / (b,c), 256 threads.
// ============================================================
__global__ void knn_kernel(const float* __restrict__ pc,
                           const float* __restrict__ centers,
                           float* __restrict__ normed) {
  const int bc = blockIdx.x;
  const int b  = bc / CC;
  const int t  = threadIdx.x;
  const float* x = pc + (size_t)b * NN * 3;
  const float* c = centers + (size_t)bc * 3;
  float cx = c[0], cy = c[1], cz = c[2];

  float dd[32];
#pragma unroll
  for (int j = 0; j < 32; ++j) {
    int p = j * 256 + t;
    float dx = x[p * 3 + 0] - cx, dy = x[p * 3 + 1] - cy, dz = x[p * 3 + 2] - cz;
    dd[j] = dx * dx + dy * dy + dz * dz;
  }

  __shared__ float rv[256];
  __shared__ int   ri[256];
  __shared__ int   selp[KK];

  for (int kk = 0; kk < KK; ++kk) {
    float bestv = 3.4e38f; int besti = 0x7fffffff;
#pragma unroll
    for (int j = 0; j < 32; ++j) {
      if (dd[j] < bestv) { bestv = dd[j]; besti = j * 256 + t; }
    }
    rv[t] = bestv; ri[t] = besti;
    __syncthreads();
    for (int s = 128; s > 0; s >>= 1) {
      if (t < s) {
        float v2 = rv[t + s]; int i2 = ri[t + s];
        if (v2 < rv[t] || (v2 == rv[t] && i2 < ri[t])) { rv[t] = v2; ri[t] = i2; }
      }
      __syncthreads();
    }
    int wsel = ri[0];
    if (t == 0) selp[kk] = wsel;
    if (t == (wsel & 255)) dd[wsel >> 8] = 3.4e38f;
    __syncthreads();
  }

  if (t < KK) {
    int p = selp[t];
    float* o = normed + ((size_t)bc * KK + t) * 3;
    o[0] = x[p * 3 + 0] - cx;
    o[1] = x[p * 3 + 1] - cy;
    o[2] = x[p * 3 + 2] - cz;
  }
}

// ============================================================
// Kernel W: transpose+convert f32 [Kd][Nd] -> bf16 [Nd][Kd]
// ============================================================
__global__ void transpose_bf16_kernel(const float* __restrict__ src,
                                      __bf16* __restrict__ dst, int Kd, int Nd) {
  int i = blockIdx.x * blockDim.x + threadIdx.x;
  if (i >= Kd * Nd) return;
  int n = i / Kd, k = i % Kd;
  dst[(size_t)n * Kd + k] = (__bf16)src[(size_t)k * Nd + n];
}

// ============================================================
// Kernel D: t = gelu(centers @ Wc1 + bc1) -> bf16 [BC][H]
// ============================================================
__global__ void center_t_kernel(const float* __restrict__ centers,
                                const float* __restrict__ Wc1,
                                const float* __restrict__ bc1,
                                __bf16* __restrict__ tA) {
  size_t i = (size_t)blockIdx.x * blockDim.x + threadIdx.x;
  if (i >= (size_t)BC * HH) return;
  int row = (int)(i / HH), n = (int)(i % HH);
  const float* c = centers + (size_t)row * 3;
  float acc = bc1[n] + c[0] * Wc1[n] + c[1] * Wc1[HH + n] + c[2] * Wc1[2 * HH + n];
  tA[i] = (__bf16)gelu_tanh(acc);
}

// ============================================================
// Kernel E: emb = t @ Wc2 + bc2  (bf16 WMMA, M=16384 N=768 K=768)
// Block tile 64x128, 256 threads = 8 waves, 4 accumulators/wave.
// Double-buffered async global->LDS staging of 64-deep K chunks.
// ============================================================
__device__ __forceinline__ void stage_chunk(__bf16* sA, __bf16* sB,
                                            const __bf16* __restrict__ tA,
                                            const __bf16* __restrict__ Wc2T,
                                            int rowblk, int nblk, int k0, int t) {
  // A tile: 64 rows x 64 k  -> 512 b128 transfers (2 / thread)
#pragma unroll
  for (int j = 0; j < 2; ++j) {
    int i = t + 256 * j;
    int row = i >> 3, seg = i & 7;
    const __bf16* g = tA + (size_t)(rowblk + row) * HH + k0 + seg * 8;
    async_b128(lds_off32(sA + row * 64 + seg * 8), g);
  }
  // B tile: 128 rows (n) x 64 k -> 1024 b128 transfers (4 / thread)
#pragma unroll
  for (int j = 0; j < 4; ++j) {
    int i = t + 256 * j;
    int row = i >> 3, seg = i & 7;
    const __bf16* g = Wc2T + (size_t)(nblk + row) * HH + k0 + seg * 8;
    async_b128(lds_off32(sB + row * 64 + seg * 8), g);
  }
}

__global__ void center_gemm_kernel(const __bf16* __restrict__ tA,
                                   const __bf16* __restrict__ Wc2T,
                                   const float* __restrict__ bc2,
                                   float* __restrict__ emb) {
  __shared__ __bf16 sA[2][64 * 64];    // 2 x 8 KB
  __shared__ __bf16 sB[2][128 * 64];   // 2 x 16 KB
  const int t = threadIdx.x, lane = t & 31, w = t >> 5;
  const int mt = w >> 1, ng = w & 1;
  const int rowblk = blockIdx.x * 64;
  const int nblk   = blockIdx.y * 128;
  const int lane16 = lane & 15, half = lane >> 4;
  const int NCH = HH / 64;             // 12 K-chunks

  stage_chunk(sA[0], sB[0], tA, Wc2T, rowblk, nblk, 0, t);
  wait_async0();
  __syncthreads();

  v8f acc[4] = {};
  for (int ch = 0; ch < NCH; ++ch) {
    int cur = ch & 1;
    if (ch + 1 < NCH)
      stage_chunk(sA[cur ^ 1], sB[cur ^ 1], tA, Wc2T, rowblk, nblk, (ch + 1) * 64, t);
#pragma unroll
    for (int ks = 0; ks < 2; ++ks) {
      v16bf a = load_a_frag(sA[cur], 64, mt * 16, ks * 32, lane);
#pragma unroll
      for (int nt = 0; nt < 4; ++nt) {
        v16bf bb = load_b_frag(sB[cur], 64, ng * 64 + nt * 16, ks * 32, lane);
        acc[nt] = __builtin_amdgcn_wmma_f32_16x16x32_bf16(
            false, a, false, bb, (short)0, acc[nt], false, false);
      }
    }
    wait_async0();      // next chunk resident in LDS
    __syncthreads();    // all waves done reading `cur` before it is re-staged
  }

#pragma unroll
  for (int nt = 0; nt < 4; ++nt) {
#pragma unroll
    for (int r = 0; r < 8; ++r) {
      int row = rowblk + mt * 16 + r + half * 8;
      int col = nblk + ng * 64 + nt * 16 + lane16;
      emb[(size_t)row * HH + col] = acc[nt][r] + bc2[col];
    }
  }
}

// ============================================================
// Kernel C: fused per-center shared-MLP chain + maxpool + W4,
// accumulated into emb (after kernel E). 1 block/(b,c), 128 thr.
// ============================================================
__global__ void knn_mlp_kernel(const float* __restrict__ normed,
                               const float* __restrict__ W1,  const float* __restrict__ b1,
                               const float* __restrict__ g1,  const float* __restrict__ be1,
                               const float* __restrict__ b2v, const float* __restrict__ g2,
                               const float* __restrict__ be2,
                               const float* __restrict__ b3v, const float* __restrict__ g3,
                               const float* __restrict__ be3,
                               const float* __restrict__ b4v,
                               const __bf16* __restrict__ W2T,
                               const __bf16* __restrict__ W3T,
                               const __bf16* __restrict__ W4T,
                               float* __restrict__ emb) {
  __shared__ float  sF[32 * 256];      // current layer f32 output
  __shared__ __bf16 sH[32 * 128];      // current layer bf16 input
  __shared__ float  s_mean[32], s_rstd[32];
  __shared__ float  pooled[256];

  const int bc = blockIdx.x, t = threadIdx.x;
  const int lane = t & 31, w = t >> 5;
  const int lane16 = lane & 15, half = lane >> 4;
  const float* nm = normed + (size_t)bc * KK * 3;

  // ---- layer1: 32x3 @ 3x64 (+b1) ----
  for (int i = t; i < 32 * 64; i += 128) {
    int r = i >> 6, c = i & 63;
    float acc = b1[c];
#pragma unroll
    for (int k2 = 0; k2 < 3; ++k2) acc += nm[r * 3 + k2] * W1[k2 * 64 + c];
    sF[r * 64 + c] = acc;
  }
  __syncthreads();
  if (t < 32) {
    float m = 0.f;
    for (int j = 0; j < 64; ++j) m += sF[t * 64 + j];
    m *= (1.f / 64.f);
    float v = 0.f;
    for (int j = 0; j < 64; ++j) { float d = sF[t * 64 + j] - m; v += d * d; }
    v *= (1.f / 64.f);
    s_mean[t] = m; s_rstd[t] = rsqrtf(v + LN_EPS);
  }
  __syncthreads();
  for (int i = t; i < 32 * 64; i += 128) {
    int r = i >> 6, c = i & 63;
    float hv = (sF[i] - s_mean[r]) * s_rstd[r] * g1[c] + be1[c];
    sH[i] = (__bf16)gelu_tanh(hv);   // sH used as [32][64]
  }
  __syncthreads();

  // ---- layer2 WMMA: 32x64 @ 64x128 ----
  for (int mt = 0; mt < 2; ++mt) {
    for (int ntl = 0; ntl < 2; ++ntl) {
      int nt = w * 2 + ntl;
      v8f acc = {};
#pragma unroll
      for (int ks = 0; ks < 2; ++ks) {
        v16bf a  = load_a_frag(sH, 64, mt * 16, ks * 32, lane);
        v16bf bb = load_b_frag(W2T, 64, nt * 16, ks * 32, lane);
        acc = __builtin_amdgcn_wmma_f32_16x16x32_bf16(
            false, a, false, bb, (short)0, acc, false, false);
      }
#pragma unroll
      for (int r = 0; r < 8; ++r) {
        int row = mt * 16 + r + half * 8;
        int col = nt * 16 + lane16;
        sF[row * 128 + col] = acc[r] + b2v[col];
      }
    }
  }
  __syncthreads();
  if (t < 32) {
    float m = 0.f;
    for (int j = 0; j < 128; ++j) m += sF[t * 128 + j];
    m *= (1.f / 128.f);
    float v = 0.f;
    for (int j = 0; j < 128; ++j) { float d = sF[t * 128 + j] - m; v += d * d; }
    v *= (1.f / 128.f);
    s_mean[t] = m; s_rstd[t] = rsqrtf(v + LN_EPS);
  }
  __syncthreads();
  for (int i = t; i < 32 * 128; i += 128) {
    int r = i >> 7, c = i & 127;
    float hv = (sF[i] - s_mean[r]) * s_rstd[r] * g2[c] + be2[c];
    sH[i] = (__bf16)gelu_tanh(hv);   // sH used as [32][128]
  }
  __syncthreads();

  // ---- layer3 WMMA: 32x128 @ 128x256 ----
  for (int mt = 0; mt < 2; ++mt) {
    for (int ntl = 0; ntl < 4; ++ntl) {
      int nt = w * 4 + ntl;
      v8f acc = {};
#pragma unroll
      for (int ks = 0; ks < 4; ++ks) {
        v16bf a  = load_a_frag(sH, 128, mt * 16, ks * 32, lane);
        v16bf bb = load_b_frag(W3T, 128, nt * 16, ks * 32, lane);
        acc = __builtin_amdgcn_wmma_f32_16x16x32_bf16(
            false, a, false, bb, (short)0, acc, false, false);
      }
#pragma unroll
      for (int r = 0; r < 8; ++r) {
        int row = mt * 16 + r + half * 8;
        int col = nt * 16 + lane16;
        sF[row * 256 + col] = acc[r] + b3v[col];
      }
    }
  }
  __syncthreads();
  if (t < 32) {
    float m = 0.f;
    for (int j = 0; j < 256; ++j) m += sF[t * 256 + j];
    m *= (1.f / 256.f);
    float v = 0.f;
    for (int j = 0; j < 256; ++j) { float d = sF[t * 256 + j] - m; v += d * d; }
    v *= (1.f / 256.f);
    s_mean[t] = m; s_rstd[t] = rsqrtf(v + LN_EPS);
  }
  __syncthreads();
  for (int i = t; i < 32 * 256; i += 128) {
    int r = i >> 8, c = i & 255;
    float hv = (sF[i] - s_mean[r]) * s_rstd[r] * g3[c] + be3[c];
    sF[i] = gelu_tanh(hv);
  }
  __syncthreads();

  // ---- maxpool over K=32 ----
  for (int c = t; c < 256; c += 128) {
    float m = sF[c];
#pragma unroll
    for (int r = 1; r < 32; ++r) m = fmaxf(m, sF[r * 256 + c]);
    pooled[c] = m;
  }
  __syncthreads();

  // ---- emb += pooled @ W4 + b4 ----
  for (int n = t; n < HH; n += 128) {
    const __bf16* wp = W4T + (size_t)n * 256;
    float acc = 0.f;
    for (int k2 = 0; k2 < 256; ++k2) acc += pooled[k2] * (float)wp[k2];
    float* e = emb + (size_t)bc * HH + n;
    *e = *e + acc + b4v[n];
  }
}

// ============================================================
// launcher
// ============================================================
extern "C" void kernel_launch(void* const* d_in, const int* in_sizes, int n_in,
                              void* d_out, int out_size, void* d_ws, size_t ws_size,
                              hipStream_t stream) {
  const float* pc  = (const float*)d_in[0];
  const float* W1  = (const float*)d_in[1];
  const float* b1  = (const float*)d_in[2];
  const float* g1  = (const float*)d_in[3];
  const float* be1 = (const float*)d_in[4];
  const float* W2  = (const float*)d_in[5];
  const float* b2  = (const float*)d_in[6];
  const float* g2  = (const float*)d_in[7];
  const float* be2 = (const float*)d_in[8];
  const float* W3  = (const float*)d_in[9];
  const float* b3  = (const float*)d_in[10];
  const float* g3  = (const float*)d_in[11];
  const float* be3 = (const float*)d_in[12];
  const float* W4  = (const float*)d_in[13];
  const float* b4  = (const float*)d_in[14];
  const float* Wc1 = (const float*)d_in[15];
  const float* bc1 = (const float*)d_in[16];
  const float* Wc2 = (const float*)d_in[17];
  const float* bc2 = (const float*)d_in[18];

  float* out     = (float*)d_out;
  float* emb     = out;                                  // [BC][768]
  float* centers = out + (size_t)BC * HH;                // [BC][3]
  float* normed  = centers + (size_t)BC * 3;             // [BC][32][3]

  char* ws = (char*)d_ws;
  size_t off = 0;
  __bf16* tA   = (__bf16*)(ws + off); off += (size_t)BC * HH * 2;    // 25.2 MB
  __bf16* W2T  = (__bf16*)(ws + off); off += (size_t)128 * 64 * 2;
  __bf16* W3T  = (__bf16*)(ws + off); off += (size_t)256 * 128 * 2;
  __bf16* W4T  = (__bf16*)(ws + off); off += (size_t)768 * 256 * 2;
  __bf16* Wc2T = (__bf16*)(ws + off); off += (size_t)768 * 768 * 2;

  fps_kernel<<<BB, 256, 0, stream>>>(pc, centers);
  knn_kernel<<<BC, 256, 0, stream>>>(pc, centers, normed);

  transpose_bf16_kernel<<<(64 * 128 + 255) / 256, 256, 0, stream>>>(W2, W2T, 64, 128);
  transpose_bf16_kernel<<<(128 * 256 + 255) / 256, 256, 0, stream>>>(W3, W3T, 128, 256);
  transpose_bf16_kernel<<<(256 * 768 + 255) / 256, 256, 0, stream>>>(W4, W4T, 256, 768);
  transpose_bf16_kernel<<<(768 * 768 + 255) / 256, 256, 0, stream>>>(Wc2, Wc2T, 768, 768);

  center_t_kernel<<<((size_t)BC * HH + 255) / 256, 256, 0, stream>>>(centers, Wc1, bc1, tA);
  center_gemm_kernel<<<dim3(BC / 64, HH / 128), 256, 0, stream>>>(tA, Wc2T, bc2, emb);

  knn_mlp_kernel<<<BC, 128, 0, stream>>>(normed, W1, b1, g1, be1,
                                         b2, g2, be2, b3, g3, be3, b4,
                                         W2T, W3T, W4T, emb);
}
